// Block_12369505812822
// MI455X (gfx1250) — matrix operand order repaired and checked by reading, not compiled, not run on hardware
//
#include <hip/hip_runtime.h>
#include <stdint.h>

// ---------------------------------------------------------------------------
// CDNA5 (gfx1250, wave32) implementation of the mixed RWKV/attention block.
// - all matmuls on v_wmma_f32_16x16x32_f16 (f16 operands, f32 accum)
// - GEMM: 8-wave WG tile 64x128, wave tile 32x32 (2x2 register blocking);
//   BOTH A and B panels staged to LDS with global_load_async_to_lds_b128
//   (double-buffered, s_wait_asynccnt + barrier) so every global byte enters
//   the WGP exactly once per workgroup; WGP-scope global_prefetch for next K.
// - attention: two-pass streaming softmax; post-softmax Toeplitz decay w and
//   Wmix head-mix fused in the apply pass via LDS (never materializes T*T).
// ---------------------------------------------------------------------------

typedef __attribute__((ext_vector_type(16))) _Float16 v16h;
typedef __attribute__((ext_vector_type(8)))  _Float16 v8h;
typedef __attribute__((ext_vector_type(8)))  float    v8f;

constexpr int Bc   = 4;
constexpr int Tc   = 1024;
constexpr int Cc   = 1024;
constexpr int Hc   = 16;
constexpr int HSc  = 64;     // head size
constexpr int ROTc = 32;     // rotary dims
constexpr int HIDc = 3072;
constexpr int Mc   = Bc * Tc;   // 4096 rows

#define DEVFN __device__ __forceinline__

DEVFN v8f wmma_f16(v16h a, v16h b, v8f c) {
  return __builtin_amdgcn_wmma_f32_16x16x32_f16(
      /*neg_a=*/false, a, /*neg_b=*/false, b,
      /*c_mod=*/(short)0, c, /*reuse_a=*/false, /*reuse_b=*/false);
}

// A fragment: 16x32 f16, row-major source (lda halves).
// lane: row = lane&15 ; kb = 16*(lane>>4); halves [kb,kb+8) then [kb+8,kb+16).
DEVFN v16h load_a_frag(const _Float16* A, int lda, int row0, int k0) {
  int lane = threadIdx.x & 31;
  const _Float16* p = A + (size_t)(row0 + (lane & 15)) * lda + k0 + ((lane >> 4) << 4);
  v8h lo = *(const v8h*)p;
  v8h hi = *(const v8h*)(p + 8);
  v16h r;
#pragma unroll
  for (int i = 0; i < 8; ++i) { r[i] = lo[i]; r[8 + i] = hi[i]; }
  return r;
}

// B fragment: 32x16 f16 from N-major (transposed) storage Bt[N][K].
// lane: col n = lane&15 ; K range = 16*(lane>>4) .. +16, contiguous.
DEVFN v16h load_bt_frag(const _Float16* Bt, int ldb, int n0, int k0) {
  int lane = threadIdx.x & 31;
  const _Float16* p = Bt + (size_t)(n0 + (lane & 15)) * ldb + k0 + ((lane >> 4) << 4);
  return *(const v16h*)p;
}

// ---------------------------------------------------------------------------
// Elementwise / prep kernels
// ---------------------------------------------------------------------------

__global__ void wconv_kernel(const float* __restrict__ W, _Float16* __restrict__ Wt,
                             int K, int N) {
  size_t idx = (size_t)blockIdx.x * blockDim.x + threadIdx.x;
  if (idx >= (size_t)K * N) return;
  int k = (int)(idx / N);
  int n = (int)(idx % N);
  Wt[(size_t)n * K + k] = (_Float16)W[idx];
}

__global__ void f2h_kernel(const float* __restrict__ src, _Float16* __restrict__ dst, size_t n) {
  size_t idx = (size_t)blockIdx.x * blockDim.x + threadIdx.x;
  if (idx < n) dst[idx] = (_Float16)src[idx];
}

__global__ void ln_kernel(const float* __restrict__ x, const float* __restrict__ g,
                          const float* __restrict__ b, _Float16* __restrict__ out) {
  int row = blockIdx.x;
  const float* xr = x + (size_t)row * Cc;
  float s = 0.f, s2 = 0.f;
  for (int c = threadIdx.x; c < Cc; c += blockDim.x) {
    float v = xr[c]; s += v; s2 += v * v;
  }
#pragma unroll
  for (int m = 16; m > 0; m >>= 1) {
    s  += __shfl_xor(s,  m, 32);
    s2 += __shfl_xor(s2, m, 32);
  }
  __shared__ float sh[16];
  __shared__ float smu, srs;
  int wv = threadIdx.x >> 5, lane = threadIdx.x & 31;
  if (lane == 0) { sh[wv] = s; sh[8 + wv] = s2; }
  __syncthreads();
  if (threadIdx.x == 0) {
    float S = 0.f, S2 = 0.f;
#pragma unroll
    for (int i = 0; i < 8; ++i) { S += sh[i]; S2 += sh[8 + i]; }
    float mu = S / (float)Cc;
    float var = S2 / (float)Cc - mu * mu;
    smu = mu; srs = rsqrtf(var + 1e-5f);
  }
  __syncthreads();
  float mu = smu, rs = srs;
  for (int c = threadIdx.x; c < Cc; c += blockDim.x)
    out[(size_t)row * Cc + c] = (_Float16)((xr[c] - mu) * rs * g[c] + b[c]);
}

__global__ void shift_kernel(const _Float16* __restrict__ h, _Float16* __restrict__ hs) {
  size_t idx = (size_t)blockIdx.x * blockDim.x + threadIdx.x;
  if (idx >= (size_t)Mc * Cc) return;
  int c = (int)(idx % Cc);
  int t = (int)((idx / Cc) % Tc);
  if (c < Cc / 2) hs[idx] = (t > 0) ? h[idx - Cc] : (_Float16)0.f;
  else            hs[idx] = h[idx];
}

__global__ void rope_pack_kernel(const float* __restrict__ qf, const float* __restrict__ kf,
                                 const float* __restrict__ vf,
                                 _Float16* __restrict__ qh, _Float16* __restrict__ kh,
                                 _Float16* __restrict__ vt) {
  size_t idx = (size_t)blockIdx.x * blockDim.x + threadIdx.x;
  if (idx >= (size_t)Mc * Cc) return;
  int d = (int)(idx % HSc);
  int t = (int)((idx / HSc) % Tc);
  int h = (int)((idx / ((size_t)HSc * Tc)) % Hc);
  int b = (int)(idx / ((size_t)HSc * Tc * Hc));
  size_t src = ((size_t)b * Tc + t) * Cc + h * HSc + d;
  float qv = qf[src], kv = kf[src], vv = vf[src];
  if (d < ROTc) {
    int i = d & (ROTc / 2 - 1);
    float theta = (float)t * __powf(10000.f, -(float)(2 * i) / (float)ROTc);
    float sn, cs;
    __sincosf(theta, &sn, &cs);
    size_t psrc = (d < ROTc / 2) ? src + ROTc / 2 : src - ROTc / 2;
    float sgn = (d < ROTc / 2) ? -1.f : 1.f;
    qv = qv * cs + sgn * qf[psrc] * sn;
    kv = kv * cs + sgn * kf[psrc] * sn;
  }
  size_t dsth = ((size_t)(b * Hc + h) * Tc + t) * HSc + d;
  qh[dsth] = (_Float16)qv;
  kh[dsth] = (_Float16)kv;
  vt[((size_t)(b * Hc + h) * HSc + d) * Tc + t] = (_Float16)vv;
}

__global__ void gelumul_kernel(const float* __restrict__ kk, const float* __restrict__ vv,
                               _Float16* __restrict__ gg, size_t n) {
  size_t idx = (size_t)blockIdx.x * blockDim.x + threadIdx.x;
  if (idx >= n) return;
  float k = kk[idx];
  float ge = 0.5f * k * (1.f + erff(k * 0.70710678118654752f));
  gg[idx] = (_Float16)(ge * vv[idx]);
}

// ---------------------------------------------------------------------------
// WMMA GEMM v3: C[M,N] = A[M,K](f16) * Bt[N,K](f16)^T (+bias, *gamma, +addsrc)
// WG = 256 threads (8 waves as 2x4), WG tile 64x128, wave tile 32x32.
// A panel (64 x 32k) AND B panel (128 x 32k) double-buffered in LDS via
// global_load_async_to_lds_b128; s_wait_asynccnt 0 + barrier each step.
// ---------------------------------------------------------------------------
__global__ void gemm_f16_v3(const _Float16* __restrict__ A, const _Float16* __restrict__ Bt,
                            const float* __restrict__ bias, float* __restrict__ Cmat,
                            int Mm, int Nn, int Kk,
                            const float* __restrict__ addsrc,
                            const float* __restrict__ gamma) {
  __shared__ __align__(16) _Float16 As[2][64][40];    // 40-half pitch (80B, 16B-mult)
  __shared__ __align__(16) _Float16 Bs[2][128][40];
  int wgN  = Nn >> 7;
  int row0 = (blockIdx.x / wgN) << 6;
  int col0 = (blockIdx.x % wgN) << 7;
  int wid  = threadIdx.x >> 5;
  int wm   = wid >> 2;          // 0..1 (M half)
  int wn   = wid & 3;           // 0..3 (N quarter)
  int lane = threadIdx.x & 31;

  // async stage mapping (16B per load):
  //   A: thread t -> row = t>>2 (0..63),  seg = t&3
  //   B: thread t -> rows t>>1? use 2 reps: idx = t + rep*256 -> row = idx>>2 (0..127)
  int arow = threadIdx.x >> 2;
  int aseg = threadIdx.x & 3;
  const _Float16* agp = A + (size_t)(row0 + arow) * Kk + aseg * 8;
  unsigned alds0 = (unsigned)(uintptr_t)(&As[0][arow][aseg * 8]);
  unsigned alds1 = (unsigned)(uintptr_t)(&As[1][arow][aseg * 8]);

  int bidx0 = threadIdx.x;          // rep 0
  int bidx1 = threadIdx.x + 256;    // rep 1
  int brow0 = bidx0 >> 2, bseg0 = bidx0 & 3;
  int brow1 = bidx1 >> 2, bseg1 = bidx1 & 3;
  const _Float16* bgp0 = Bt + (size_t)(col0 + brow0) * Kk + bseg0 * 8;
  const _Float16* bgp1 = Bt + (size_t)(col0 + brow1) * Kk + bseg1 * 8;
  unsigned blds0a = (unsigned)(uintptr_t)(&Bs[0][brow0][bseg0 * 8]);
  unsigned blds0b = (unsigned)(uintptr_t)(&Bs[0][brow1][bseg1 * 8]);
  unsigned blds1a = (unsigned)(uintptr_t)(&Bs[1][brow0][bseg0 * 8]);
  unsigned blds1b = (unsigned)(uintptr_t)(&Bs[1][brow1][bseg1 * 8]);

  asm volatile("global_load_async_to_lds_b128 %0, %1, off"
               :: "v"(alds0), "v"(agp) : "memory");
  asm volatile("global_load_async_to_lds_b128 %0, %1, off"
               :: "v"(blds0a), "v"(bgp0) : "memory");
  asm volatile("global_load_async_to_lds_b128 %0, %1, off"
               :: "v"(blds0b), "v"(bgp1) : "memory");
  asm volatile("s_wait_asynccnt 0" ::: "memory");
  __syncthreads();

  v8f acc[2][2];
#pragma unroll
  for (int mt = 0; mt < 2; ++mt)
#pragma unroll
    for (int nt = 0; nt < 2; ++nt) acc[mt][nt] = (v8f){};

  int nsteps = Kk >> 5;
  for (int ki = 0; ki < nsteps; ++ki) {
    int cur = ki & 1;
    if (ki + 1 < nsteps) {
      size_t kadv = (size_t)(ki + 1) * 32;
      asm volatile("global_load_async_to_lds_b128 %0, %1, off"
                   :: "v"(cur ? alds0 : alds1), "v"(agp + kadv) : "memory");
      asm volatile("global_load_async_to_lds_b128 %0, %1, off"
                   :: "v"(cur ? blds0a : blds1a), "v"(bgp0 + kadv) : "memory");
      asm volatile("global_load_async_to_lds_b128 %0, %1, off"
                   :: "v"(cur ? blds0b : blds1b), "v"(bgp1 + kadv) : "memory");
      if (ki + 2 < nsteps)   // warm WGP cache for the K-chunk after next
        __builtin_prefetch(agp + (size_t)(ki + 2) * 32, 0, 3);
    }
    // A fragments from LDS (shared across the 4 N-waves)
    v16h af[2];
#pragma unroll
    for (int mt = 0; mt < 2; ++mt) {
      const _Float16* lp = &As[cur][(wm << 5) + (mt << 4) + (lane & 15)][(lane >> 4) << 4];
      v8h lo = *(const v8h*)lp;
      v8h hi = *(const v8h*)(lp + 8);
      v16h r;
#pragma unroll
      for (int i = 0; i < 8; ++i) { r[i] = lo[i]; r[8 + i] = hi[i]; }
      af[mt] = r;
    }
    // B fragments from LDS (shared across the 2 M-waves)
    v16h bf[2];
#pragma unroll
    for (int nt = 0; nt < 2; ++nt) {
      const _Float16* lp = &Bs[cur][(wn << 5) + (nt << 4) + (lane & 15)][(lane >> 4) << 4];
      bf[nt] = *(const v16h*)lp;
    }

#pragma unroll
    for (int mt = 0; mt < 2; ++mt)
#pragma unroll
      for (int nt = 0; nt < 2; ++nt)
        acc[mt][nt] = wmma_f16(af[mt], bf[nt], acc[mt][nt]);

    if (ki + 1 < nsteps)
      asm volatile("s_wait_asynccnt 0" ::: "memory");
    __syncthreads();
  }

#pragma unroll
  for (int mt = 0; mt < 2; ++mt) {
    int rbase = row0 + (wm << 5) + (mt << 4) + ((lane >> 4) << 3);
#pragma unroll
    for (int nt = 0; nt < 2; ++nt) {
      int n = col0 + (wn << 5) + (nt << 4) + (lane & 15);
      float bn = bias ? bias[n] : 0.f;
#pragma unroll
      for (int p = 0; p < 8; ++p) {
        int row = rbase + p;
        float v = acc[mt][nt][p] + bn;
        if (gamma)  v *= gamma[row & (Tc - 1)];
        if (addsrc) v += addsrc[(size_t)row * Nn + n];
        Cmat[(size_t)row * Nn + n] = v;
      }
    }
  }
}

// ---------------------------------------------------------------------------
// Attention pass A: per (b,h,t-tile) softmax stats m,l over causal columns.
// ---------------------------------------------------------------------------
__global__ void attn_stats_kernel(const _Float16* __restrict__ qh, const _Float16* __restrict__ kh,
                                  float* __restrict__ mout, float* __restrict__ lout) {
  const int tilesT = Tc / 16;
  int wid = blockIdx.x * (blockDim.x >> 5) + (threadIdx.x >> 5);
  int bh = wid / tilesT;
  int t0 = (wid % tilesT) << 4;
  if (bh >= Bc * Hc) return;
  int lane = threadIdx.x & 31;
  int hi = lane >> 4;
  const _Float16* qb = qh + (size_t)bh * Tc * HSc;
  const _Float16* kb = kh + (size_t)bh * Tc * HSc;
  v16h a0 = load_a_frag(qb, HSc, t0, 0);
  v16h a1 = load_a_frag(qb, HSc, t0, 32);
  float rm[8], rl[8];
#pragma unroll
  for (int p = 0; p < 8; ++p) { rm[p] = -3.0e38f; rl[p] = 0.f; }
  for (int u0 = 0; u0 <= t0; u0 += 16) {
    v8f acc = {};
    acc = wmma_f16(a0, load_bt_frag(kb, HSc, u0, 0),  acc);
    acc = wmma_f16(a1, load_bt_frag(kb, HSc, u0, 32), acc);
    int col = u0 + (lane & 15);
#pragma unroll
    for (int p = 0; p < 8; ++p) {
      int row = t0 + p + (hi << 3);
      float s = acc[p] * 0.125f;                 // 1/sqrt(HS)
      if (col > row) s = -3.0e38f;
      float tm = s;
#pragma unroll
      for (int mm = 8; mm > 0; mm >>= 1) tm = fmaxf(tm, __shfl_xor(tm, mm, 16));
      float nm = fmaxf(rm[p], tm);
      float e = __expf(s - nm);
      float ts = e;
#pragma unroll
      for (int mm = 8; mm > 0; mm >>= 1) ts += __shfl_xor(ts, mm, 16);
      rl[p] = rl[p] * __expf(rm[p] - nm) + ts;
      rm[p] = nm;
    }
  }
  if ((lane & 15) == 0) {
#pragma unroll
    for (int p = 0; p < 8; ++p) {
      int row = t0 + p + (hi << 3);
      mout[(size_t)bh * Tc + row] = rm[p];
      lout[(size_t)bh * Tc + row] = rl[p];
    }
  }
}

// ---------------------------------------------------------------------------
// Attention pass B: one workgroup (16 waves = all heads) per (b, t-tile).
// ---------------------------------------------------------------------------
__global__ void attn_apply_kernel(const _Float16* __restrict__ qh, const _Float16* __restrict__ kh,
                                  const _Float16* __restrict__ vt,
                                  const float* __restrict__ mstat, const float* __restrict__ lstat,
                                  const float* __restrict__ time_w, const float* __restrict__ alpha,
                                  const float* __restrict__ beta,  const float* __restrict__ wmix,
                                  float* __restrict__ ybuf) {
  const int tilesT = Tc / 16;
  int b  = blockIdx.x / tilesT;
  int t0 = (blockIdx.x % tilesT) << 4;
  int wv = threadIdx.x >> 5;        // head index (score role j == output role i)
  int lane = threadIdx.x & 31;
  int hi = lane >> 4;

  __shared__ float Pl[16][16][32];
  __shared__ float wmix_s[Hc * Hc];
  if (threadIdx.x < Hc * Hc) wmix_s[threadIdx.x] = wmix[threadIdx.x];
  __syncthreads();

  int bh = b * Hc + wv;
  const _Float16* qb = qh + (size_t)bh * Tc * HSc;
  const _Float16* kb = kh + (size_t)bh * Tc * HSc;
  const _Float16* vb = vt + (size_t)bh * HSc * Tc;

  v16h a0 = load_a_frag(qb, HSc, t0, 0);
  v16h a1 = load_a_frag(qb, HSc, t0, 32);

  float mrow[8], lrow[8], brow[8];
#pragma unroll
  for (int p = 0; p < 8; ++p) {
    int row = t0 + p + (hi << 3);
    mrow[p] = mstat[(size_t)bh * Tc + row];
    lrow[p] = lstat[(size_t)bh * Tc + row];
    brow[p] = beta[(size_t)wv * Tc + row];
  }

  v8f acc[4];
#pragma unroll
  for (int d4 = 0; d4 < 4; ++d4) acc[d4] = (v8f){};

  for (int u0 = 0; u0 <= t0 + 15; u0 += 32) {
    v8f s0 = {}, s1 = {};
    s0 = wmma_f16(a0, load_bt_frag(kb, HSc, u0,      0),  s0);
    s0 = wmma_f16(a1, load_bt_frag(kb, HSc, u0,      32), s0);
    s1 = wmma_f16(a0, load_bt_frag(kb, HSc, u0 + 16, 0),  s1);
    s1 = wmma_f16(a1, load_bt_frag(kb, HSc, u0 + 16, 32), s1);
#pragma unroll
    for (int sub = 0; sub < 2; ++sub) {
      int col = u0 + sub * 16 + (lane & 15);
      float al = alpha[(size_t)wv * Tc + col];
#pragma unroll
      for (int p = 0; p < 8; ++p) {
        int row = t0 + p + (hi << 3);
        float sv = (sub == 0) ? s0[p] : s1[p];
        float pv = 0.f;
        if (col <= row) {
          float e  = __expf(sv * 0.125f - mrow[p]);
          float tw = time_w[(size_t)wv * Tc + (Tc - 1 - (row - col))];
          pv = e / lrow[p] * tw * al * brow[p];
        }
        Pl[wv][p + (hi << 3)][sub * 16 + (lane & 15)] = pv;
      }
    }
    __syncthreads();

    // Wmix head-mix, gathered straight into A-fragment layout for head wv
    v16h afrag;
    int r = lane & 15;
    int kb8 = hi << 4;
#pragma unroll
    for (int hh = 0; hh < 2; ++hh) {
#pragma unroll
      for (int kk = 0; kk < 8; ++kk) {
        int k = kb8 + hh * 8 + kk;
        float s = 0.f;
#pragma unroll
        for (int j = 0; j < 16; ++j) s += wmix_s[wv * 16 + j] * Pl[j][r][k];
        afrag[hh * 8 + kk] = (_Float16)s;
      }
    }

    int ku = u0 + (hi << 4);
#pragma unroll
    for (int d4 = 0; d4 < 4; ++d4) {
      const _Float16* p = vb + (size_t)(d4 * 16 + (lane & 15)) * Tc + ku;
      acc[d4] = wmma_f16(afrag, *(const v16h*)p, acc[d4]);
    }
    __syncthreads();
  }

  float* yb = ybuf + ((size_t)b * Tc + t0) * Cc + wv * HSc;
#pragma unroll
  for (int d4 = 0; d4 < 4; ++d4) {
#pragma unroll
    for (int p = 0; p < 8; ++p)
      yb[(size_t)(p + (hi << 3)) * Cc + d4 * 16 + (lane & 15)] = acc[d4][p];
  }
}

// ---------------------------------------------------------------------------
// Host-side orchestration
// ---------------------------------------------------------------------------
extern "C" void kernel_launch(void* const* d_in, const int* in_sizes, int n_in,
                              void* d_out, int out_size, void* d_ws, size_t ws_size,
                              hipStream_t stream) {
  (void)in_sizes; (void)n_in; (void)out_size; (void)ws_size;
  const float* x     = (const float*)d_in[0];
  const float* g1    = (const float*)d_in[1];
  const float* b1    = (const float*)d_in[2];
  const float* g2    = (const float*)d_in[3];
  const float* b2    = (const float*)d_in[4];
  const float* Wq    = (const float*)d_in[5];
  const float* bq    = (const float*)d_in[6];
  const float* Wk    = (const float*)d_in[7];
  const float* bk    = (const float*)d_in[8];
  const float* Wv    = (const float*)d_in[9];
  const float* bv    = (const float*)d_in[10];
  const float* tw    = (const float*)d_in[11];
  const float* talpha= (const float*)d_in[12];
  const float* tbeta = (const float*)d_in[13];
  const float* tgamma= (const float*)d_in[14];
  const float* Wmix  = (const float*)d_in[15];
  const float* Wo    = (const float*)d_in[16];
  const float* bo    = (const float*)d_in[17];
  const float* Wk2   = (const float*)d_in[18];
  const float* bk2   = (const float*)d_in[19];
  const float* Wv2   = (const float*)d_in[20];
  const float* bv2   = (const float*)d_in[21];
  const float* Ww2   = (const float*)d_in[22];
  const float* bw2   = (const float*)d_in[23];
  float* out = (float*)d_out;

  char* ws = (char*)d_ws;
  size_t off = 0;
  auto alloc = [&](size_t bytes) -> size_t {
    size_t o = off;
    off += (bytes + 255) & ~(size_t)255;
    return o;
  };
  const size_t MCf = (size_t)Mc * Cc * sizeof(float);
  const size_t MCh = (size_t)Mc * Cc * sizeof(_Float16);

  size_t o_hh   = alloc(MCh);                    // h (f16); reused for h2
  size_t o_hsh  = alloc(MCh);                    // hs (f16); reused for y_h
  size_t o_qf   = alloc(MCf);                    // q f32 | start of kk reuse
  size_t o_kf   = alloc(MCf);
  size_t o_vf   = alloc(MCf);
  size_t o_qh   = alloc(MCh);                    // packed q | start of gg reuse
  size_t o_kh   = alloc(MCh);
  size_t o_vt   = alloc(MCh);
  size_t o_m    = alloc((size_t)Bc * Hc * Tc * sizeof(float));
  size_t o_l    = alloc((size_t)Bc * Hc * Tc * sizeof(float));
  size_t o_y    = alloc(MCf);
  size_t o_x1   = alloc(MCf);
  size_t o_WqT  = alloc((size_t)Cc * Cc * sizeof(_Float16));
  size_t o_WkT  = alloc((size_t)Cc * Cc * sizeof(_Float16));
  size_t o_WvT  = alloc((size_t)Cc * Cc * sizeof(_Float16));
  size_t o_WoT  = alloc((size_t)Cc * Cc * sizeof(_Float16));
  size_t o_Wk2T = alloc((size_t)Cc * HIDc * sizeof(_Float16));
  size_t o_Wv2T = alloc((size_t)Cc * HIDc * sizeof(_Float16));
  size_t o_Ww2T = alloc((size_t)HIDc * Cc * sizeof(_Float16));
  size_t o_vv   = alloc((size_t)Mc * HIDc * sizeof(float));
  size_t o_kk   = o_qf;   // q/k/v f32 region == exactly M*HID*4 bytes
  size_t o_gg   = o_qh;   // packed q/k/v f16 region == exactly M*HID*2 bytes
  size_t o_yh   = o_hsh;
  size_t o_h2   = o_hh;

  auto H16 = [&](size_t o) { return (_Float16*)(ws + o); };
  auto F32 = [&](size_t o) { return (float*)(ws + o); };

  const int TPB = 256;
  auto eb = [&](size_t n) { return (unsigned)((n + TPB - 1) / TPB); };

  // 1) weight convert + transpose (f32 KxN -> f16 NxK)
  wconv_kernel<<<eb((size_t)Cc * Cc), TPB, 0, stream>>>(Wq,  H16(o_WqT),  Cc, Cc);
  wconv_kernel<<<eb((size_t)Cc * Cc), TPB, 0, stream>>>(Wk,  H16(o_WkT),  Cc, Cc);
  wconv_kernel<<<eb((size_t)Cc * Cc), TPB, 0, stream>>>(Wv,  H16(o_WvT),  Cc, Cc);
  wconv_kernel<<<eb((size_t)Cc * Cc), TPB, 0, stream>>>(Wo,  H16(o_WoT),  Cc, Cc);
  wconv_kernel<<<eb((size_t)Cc * HIDc), TPB, 0, stream>>>(Wk2, H16(o_Wk2T), Cc, HIDc);
  wconv_kernel<<<eb((size_t)Cc * HIDc), TPB, 0, stream>>>(Wv2, H16(o_Wv2T), Cc, HIDc);
  wconv_kernel<<<eb((size_t)HIDc * Cc), TPB, 0, stream>>>(Ww2, H16(o_Ww2T), HIDc, Cc);

  // 2) LN1 + time-shift
  ln_kernel<<<Mc, TPB, 0, stream>>>(x, g1, b1, H16(o_hh));
  shift_kernel<<<eb((size_t)Mc * Cc), TPB, 0, stream>>>(H16(o_hh), H16(o_hsh));

  // 3..) GEMMs: WG tile 64x128
  auto gemm = [&](const _Float16* A, size_t oBt, const float* bias, float* C,
                  int M, int N, int K, const float* addsrc, const float* gamma) {
    unsigned blocks = (unsigned)((M >> 6) * (N >> 7));
    gemm_f16_v3<<<blocks, 256, 0, stream>>>(A, H16(oBt), bias, C, M, N, K, addsrc, gamma);
  };
  gemm(H16(o_hsh), o_WqT, bq, F32(o_qf), Mc, Cc, Cc, nullptr, nullptr);
  gemm(H16(o_hsh), o_WkT, bk, F32(o_kf), Mc, Cc, Cc, nullptr, nullptr);
  gemm(H16(o_hsh), o_WvT, bv, F32(o_vf), Mc, Cc, Cc, nullptr, nullptr);

  // 4) RoPE + pack to head-major f16
  rope_pack_kernel<<<eb((size_t)Mc * Cc), TPB, 0, stream>>>(
      F32(o_qf), F32(o_kf), F32(o_vf), H16(o_qh), H16(o_kh), H16(o_vt));

  // 5) attention pass A (softmax stats)
  {
    int waves = Bc * Hc * (Tc / 16);
    attn_stats_kernel<<<(waves + 7) / 8, 256, 0, stream>>>(H16(o_qh), H16(o_kh),
                                                           F32(o_m), F32(o_l));
  }
  // 6) attention pass B
  attn_apply_kernel<<<Bc * (Tc / 16), 512, 0, stream>>>(
      H16(o_qh), H16(o_kh), H16(o_vt), F32(o_m), F32(o_l),
      tw, talpha, tbeta, Wmix, F32(o_y));

  // 7) output proj + gamma + residual -> x1
  f2h_kernel<<<eb((size_t)Mc * Cc), TPB, 0, stream>>>(F32(o_y), H16(o_yh), (size_t)Mc * Cc);
  gemm(H16(o_yh), o_WoT, bo, F32(o_x1), Mc, Cc, Cc, x, tgamma);

  // 8) FFN
  ln_kernel<<<Mc, TPB, 0, stream>>>(F32(o_x1), g2, b2, H16(o_h2));
  gemm(H16(o_h2), o_Wk2T, bk2, F32(o_kk), Mc, HIDc, Cc, nullptr, nullptr);
  gemm(H16(o_h2), o_Wv2T, bv2, F32(o_vv), Mc, HIDc, Cc, nullptr, nullptr);
  gelumul_kernel<<<eb((size_t)Mc * HIDc), TPB, 0, stream>>>(
      F32(o_kk), F32(o_vv), H16(o_gg), (size_t)Mc * HIDc);
  gemm(H16(o_gg), o_Ww2T, bw2, out, Mc, Cc, HIDc, F32(o_x1), nullptr);
}
